// Prototype3ContrastLoss_54417235640826
// MI455X (gfx1250) — matrix operand order repaired and checked by reading, not compiled, not run on hardware
//
#include <hip/hip_runtime.h>
#include <hip/hip_bf16.h>
#include <math.h>

typedef __attribute__((ext_vector_type(16))) _Float16 v16h;
typedef __attribute__((ext_vector_type(8)))  _Float16 v8h;
typedef __attribute__((ext_vector_type(8)))  float    v8f;

#define BATCH   16
#define CCH     256
#define KCLS    21
#define KPAD    32
#define HW      16384          // 128*128
#define TILE    256            // pixels per workgroup
#define FSTR    264            // feat LDS row stride (halves): TILE + 8 pad
#define PSTR    264            // pos  LDS row stride (halves)
#define TAUV    0.07f

// ---------------- workspace zero ----------------
__global__ __launch_bounds__(256) void zero_ws_kernel(float* ws) {
    int i = blockIdx.x * 256 + threadIdx.x;
    if (i < KPAD * CCH + 64) ws[i] = 0.0f;   // protos [32*256] + counts (bit-zero ints)
}

// ---------------- main: masked-normalized reduction via WMMA ----------------
// LDS budget: 135168 (feat f16) + 21504 (gt/pos alias) + 2048 + 1024 + 128 = 159872 B
// => 2 workgroups per WGP (319744 <= 320K).
__global__ __launch_bounds__(512)
void proto_accum_kernel(const float* __restrict__ feat, const int* __restrict__ gt,
                        float* __restrict__ protos, int* __restrict__ counts) {
    __shared__ __align__(16) _Float16      s_feat[CCH * FSTR];       // feat tile, f16
    __shared__ __align__(16) unsigned char s_alias[KCLS * TILE * 4]; // gt ints, later pos f16
    __shared__ float s_ss[2][TILE];
    __shared__ float s_rn[TILE];
    __shared__ int   s_cnt[KPAD];

    int*      s_gt  = (int*)s_alias;        // [21][256] i32 (async-load destination)
    _Float16* s_pos = (_Float16*)s_alias;   // [32][PSTR] f16 (built after handoff)

    const int tid = threadIdx.x;
    const int b   = blockIdx.y;
    const int hw0 = blockIdx.x * TILE;

    // ---- stage 0: async-copy the gt tile (21 rows x 256 px) global -> LDS ----
    {
        const int* gbase = gt + (size_t)b * KCLS * HW + hw0;   // uniform
        const unsigned lds_base = (unsigned)(size_t)(&s_alias[0]);
        #pragma unroll
        for (int j = 0; j < 3; ++j) {                 // 21*64 = 1344 b128 transfers
            int idx = tid + j * 512;
            if (idx < KCLS * (TILE / 4)) {
                int k   = idx >> 6;                   // row (64 x b128 per row)
                int off = idx & 63;
                unsigned voff    = (unsigned)((size_t)k * HW * 4 + (size_t)off * 16);
                unsigned ldsaddr = lds_base + (unsigned)(idx * 16);
                asm volatile("global_load_async_to_lds_b128 %0, %1, %2"
                             :: "v"(ldsaddr), "v"(voff), "s"(gbase) : "memory");
            }
        }
    }

    // ---- pass 1: per-pixel sumsq over 256 channels; stage feat tile to LDS f16 ----
    {
        const int p    = tid & (TILE - 1);
        const int half = tid >> 8;                    // 0/1 -> channels [0,128)/[128,256)
        const float* fp = feat + ((size_t)(b * CCH + half * 128)) * HW + hw0 + p;
        _Float16* sp = &s_feat[(half * 128) * FSTR + p];
        float acc = 0.0f;
        #pragma unroll 8
        for (int c = 0; c < 128; ++c) {
            float v = *fp;
            acc += v * v;
            *sp = (_Float16)v;
            fp += HW; sp += FSTR;
        }
        s_ss[half][p] = acc;
    }
    if (tid < KPAD) s_cnt[tid] = 0;
    __syncthreads();
    if (tid < TILE) {
        float n = sqrtf(s_ss[0][tid] + s_ss[1][tid]);
        s_rn[tid] = 1.0f / fmaxf(n, 1e-12f);
    }

    // ---- wait for gt async copy, then handoff gt(regs) -> pos_scaled(LDS) ----
#if __has_builtin(__builtin_amdgcn_s_wait_asynccnt)
    __builtin_amdgcn_s_wait_asynccnt(0);
#else
    asm volatile("s_wait_asynccnt 0x0" ::: "memory");
#endif
    __syncthreads();                                  // s_rn + all gt data visible

    int gv[11];
    #pragma unroll
    for (int j = 0; j < 11; ++j) {                    // read all gt values into regs
        int idx = tid + j * 512;
        gv[j] = (idx < KCLS * TILE) ? s_gt[idx] : 0;
    }
    __syncthreads();                                  // all reads done before alias overwrite
    #pragma unroll
    for (int j = 0; j < 16; ++j) {                    // write pos_scaled (32 rows, pad w/ 0)
        int idx = tid + j * 512;
        int k = idx >> 8;
        int p = idx & 255;
        _Float16 v = (_Float16)0.0f;
        if (idx < KCLS * TILE && gv[j] == 1) {
            v = (_Float16)s_rn[p];
            atomicAdd(&s_cnt[k], 1);
        }
        s_pos[k * PSTR + p] = v;
    }
    __syncthreads();
    if (tid < KCLS) atomicAdd(&counts[tid], s_cnt[tid]);

    // ---- pass 2: WMMA from LDS only.  D[16 ch x 16 cls] += A(feat) x B(mask) ----
    const int wave = tid >> 5;              // 16 waves -> channel tile
    const int lane = tid & 31;
    const int lr   = lane & 15;
    const int hi   = lane >> 4;
    const int c0   = wave * 16;

    // A-layout: lanes 0-15 row M=lr, K = {0..7,16..23}; lanes 16-31 same rows, K+8
    const _Float16* abase = &s_feat[(c0 + lr) * FSTR + hi * 8];

    v8f d0 = {}; v8f d1 = {};
    #pragma unroll
    for (int s = 0; s < TILE / 32; ++s) {
        const int p0 = s * 32;
        v8h a0 = *(const v8h*)(abase + p0);           // K = p0 + hi*8 + 0..7
        v8h a1 = *(const v8h*)(abase + p0 + 16);      // K = p0 + 16 + hi*8 + 0..7
        v16h a = {};
        #pragma unroll
        for (int j = 0; j < 8; ++j) { a[j] = a0[j]; a[8 + j] = a1[j]; }

        // B-layout: lane n holds col N=lr, K = p0 + hi*8 + {0..7, 16..23}
        const _Float16* bp0 = &s_pos[lr * PSTR        + p0 + hi * 8];
        const _Float16* bp1 = &s_pos[(16 + lr) * PSTR + p0 + hi * 8];
        v8h b0lo = *(const v8h*)bp0;  v8h b0hi = *(const v8h*)(bp0 + 16);
        v8h b1lo = *(const v8h*)bp1;  v8h b1hi = *(const v8h*)(bp1 + 16);
        v16h b0 = {}, b1 = {};
        #pragma unroll
        for (int j = 0; j < 8; ++j) {
            b0[j] = b0lo[j]; b0[8 + j] = b0hi[j];
            b1[j] = b1lo[j]; b1[8 + j] = b1hi[j];
        }
        d0 = __builtin_amdgcn_wmma_f32_16x16x32_f16(false, a, false, b0, (short)0, d0, false, false);
        d1 = __builtin_amdgcn_wmma_f32_16x16x32_f16(false, a, false, b1, (short)0, d1, false, false);
    }

    // D-layout: lane col N=lr (class), VGPR r -> row M = hi*8 + r (channel)
    #pragma unroll
    for (int r = 0; r < 8; ++r) {
        int c = c0 + hi * 8 + r;
        atomicAdd(&protos[lr * CCH + c], d0[r]);                 // classes 0..15
        int k1 = 16 + lr;
        if (k1 < KCLS) atomicAdd(&protos[k1 * CCH + c], d1[r]);  // classes 16..20
    }
}

// ---------------- finisher: normalize, sim, log-softmax, loss ----------------
__global__ __launch_bounds__(512)
void finalize_kernel(const float* __restrict__ protos, const int* __restrict__ counts,
                     float* __restrict__ out) {
    __shared__ float P[KCLS][CCH];
    __shared__ float rinv[KCLS];
    __shared__ float sim[KCLS * KCLS];
    __shared__ float lpd[KCLS];
    __shared__ float exs[KCLS];
    const int tid = threadIdx.x;

    for (int i = tid; i < KCLS * CCH; i += 512) P[i >> 8][i & 255] = protos[i];
    __syncthreads();

    if (tid < KCLS) {
        float s = 0.0f;
        for (int c = 0; c < CCH; ++c) { float v = P[tid][c]; s += v * v; }
        rinv[tid] = 1.0f / fmaxf(sqrtf(s), 1e-12f);
        exs[tid]  = (counts[tid] > 0) ? 1.0f : 0.0f;
    }
    __syncthreads();

    if (tid < KCLS * KCLS) {
        int i = tid / KCLS, j = tid % KCLS;
        float s = 0.0f;
        for (int c = 0; c < CCH; ++c) s += P[i][c] * P[j][c];
        sim[tid] = s * rinv[i] * rinv[j] * (1.0f / TAUV);
    }
    __syncthreads();

    if (tid < KCLS) {
        // log_softmax over axis 0 (rows), evaluated at the diagonal of column tid
        float m = -INFINITY;
        for (int j = 0; j < KCLS; ++j) m = fmaxf(m, sim[j * KCLS + tid]);
        float se = 0.0f;
        for (int j = 0; j < KCLS; ++j) se += __expf(sim[j * KCLS + tid] - m);
        lpd[tid] = sim[tid * KCLS + tid] - (m + __logf(se));
    }
    __syncthreads();

    if (tid == 0) {
        float num = 0.0f, den = 0.0f;
        for (int k = 0; k < KCLS; ++k) { num += lpd[k] * exs[k]; den += exs[k]; }
        out[0] = -num / den;
    }
}

extern "C" void kernel_launch(void* const* d_in, const int* in_sizes, int n_in,
                              void* d_out, int out_size, void* d_ws, size_t ws_size,
                              hipStream_t stream) {
    const float* feat = (const float*)d_in[0];   // [16,256,128,128] f32
    const int*   gt   = (const int*)d_in[1];     // [16,21,128,128] i32
    float* protos = (float*)d_ws;                         // [32*256] f32
    int*   counts = (int*)((float*)d_ws + KPAD * CCH);    // [32] i32
    float* out    = (float*)d_out;

    zero_ws_kernel<<<dim3((KPAD * CCH + 64 + 255) / 256), 256, 0, stream>>>((float*)d_ws);
    proto_accum_kernel<<<dim3(HW / TILE, BATCH), 512, 0, stream>>>(feat, gt, protos, counts);
    finalize_kernel<<<1, 512, 0, stream>>>(protos, counts, out);
}
